// ARTModel_31645319037393
// MI455X (gfx1250) — matrix operand order repaired and checked by reading, not compiled, not run on hardware
//
#include <hip/hip_runtime.h>

// ---------- vector types (clang ext_vector: trivially constructible, SROA-friendly) ----
typedef __attribute__((ext_vector_type(16))) _Float16 v16h;
typedef __attribute__((ext_vector_type(8)))  _Float16 v8h;
typedef __attribute__((ext_vector_type(8)))  float    v8f;
typedef __attribute__((ext_vector_type(4)))  int      i4;

__device__ __forceinline__ v16h concat8(v8h lo, v8h hi) {
  return __builtin_shufflevector(lo, hi, 0, 1, 2, 3, 4, 5, 6, 7,
                                 8, 9, 10, 11, 12, 13, 14, 15);
}

#define B_    256
#define T_    2048
#define V_    64
#define D_    50
#define H_    50
#define HID_  100

// ---------- workspace layout (bytes) ----------
#define WS_EXP  0          // 64x64 f32 fused (embed@Wx + b_rnn), padded  = 16384 B
#define WS_WHS  16384      // Wh  f16, B-layout swizzled, 8 groups        =  8192 B
#define WS_W1S  24576      // W1  f16, B-layout swizzled, 16 groups       = 16384 B
#define WS_W2S  40960      // W2  f16, B-layout swizzled, 16 groups       = 16384 B
#define WS_B1   57344      // b1 padded to 128 f32                        =   512 B
#define WS_B2   57856      // b2 64 f32                                   =   256 B
#define WS_HS   65536      // hs: [B*T][64] f16                           = 64 MiB

// Branch-free tanh: CDNA5 hardware transcendental (v_tanh_f32) when available.
__device__ __forceinline__ float fast_tanh(float x) {
#if __has_builtin(__builtin_amdgcn_tanhf)
  return __builtin_amdgcn_tanhf(x);
#else
  const float xc = fminf(fmaxf(x, -20.f), 20.f);
  const float e  = __builtin_amdgcn_exp2f(xc * 2.88539008177792681472f); // 2*log2(e)
  const float r  = __builtin_amdgcn_rcpf(e + 1.0f);
  return __builtin_fmaf(-2.0f, r, 1.0f);
#endif
}

// LDS-only split barrier: waits DScnt only (global loads stay outstanding across it),
// then CDNA5 split signal/wait on the workgroup barrier (-1).
__device__ __forceinline__ void lds_barrier() {
  asm volatile("s_wait_dscnt 0x0\n\t"
               "s_barrier_signal -1\n\t"
               "s_barrier_wait -1" ::: "memory");
}

// =====================================================================================
// Kernel A: fold embedding+input-projection into a 64x64 table; swizzle weights into
// per-lane-contiguous WMMA B-layout (group g, lane l -> 32 bytes at (g*32+l)*32).
// B-layout for 16x16x32 f16: lanes 0-15 hold K=0..7,16..23, lanes 16-31 hold K=8..15,24..31,
// VGPR v holds the packed pair (K=k, K=k+1), k = 8*(lane>>4) + 2v (+8 if v>=4), for column N = lane&15.
// =====================================================================================
__global__ __launch_bounds__(256) void prep_weights(
    const float* __restrict__ embed, const float* __restrict__ Wx,
    const float* __restrict__ Wh,    const float* __restrict__ b_rnn,
    const float* __restrict__ W1,    const float* __restrict__ b1,
    const float* __restrict__ W2,    const float* __restrict__ b2,
    unsigned char* __restrict__ ws)
{
  float*    Exp = (float*)(ws + WS_EXP);
  _Float16* Whs = (_Float16*)(ws + WS_WHS);
  _Float16* W1s = (_Float16*)(ws + WS_W1S);
  _Float16* W2s = (_Float16*)(ws + WS_W2S);
  float*    b1p = (float*)(ws + WS_B1);
  float*    b2p = (float*)(ws + WS_B2);
  const int tid = threadIdx.x;

  // Exp[v][h] = sum_d embed[v][d] * Wx[d][h] + b_rnn[h]   (zero-padded h>=50)
  for (int i = tid; i < 64 * 64; i += 256) {
    const int v = i >> 6, h = i & 63;
    float s = 0.f;
    if (h < H_) {
      for (int d = 0; d < D_; ++d) s += embed[v * D_ + d] * Wx[d * H_ + h];
      s += b_rnn[h];
    }
    Exp[i] = s;
  }

  // Wh -> 8 groups (kc:2 x nt:4), padded 50x50 -> 64x64
  for (int i = tid; i < 8 * 32 * 8; i += 256) {
    const int g = i >> 8, l = (i >> 3) & 31, vv = i & 7;
    const int kc = g >> 2, nt = g & 3;
    const int n  = nt * 16 + (l & 15);
    const int k0 = kc * 32 + ((vv >> 2) << 4) + ((l >> 4) << 3) + 2 * (vv & 3);
    const float a = (k0     < H_ && n < H_) ? Wh[k0 * H_ + n]       : 0.f;
    const float b = (k0 + 1 < H_ && n < H_) ? Wh[(k0 + 1) * H_ + n] : 0.f;
    Whs[(g * 32 + l) * 16 + 2 * vv]     = (_Float16)a;
    Whs[(g * 32 + l) * 16 + 2 * vv + 1] = (_Float16)b;
  }

  // W1 -> 16 groups (kc:2 x nt:8), padded 50x100 -> 64x128
  for (int i = tid; i < 16 * 32 * 8; i += 256) {
    const int g = i >> 8, l = (i >> 3) & 31, vv = i & 7;
    const int kc = g >> 3, nt = g & 7;
    const int n  = nt * 16 + (l & 15);
    const int k0 = kc * 32 + ((vv >> 2) << 4) + ((l >> 4) << 3) + 2 * (vv & 3);
    const float a = (k0     < H_ && n < HID_) ? W1[k0 * HID_ + n]       : 0.f;
    const float b = (k0 + 1 < H_ && n < HID_) ? W1[(k0 + 1) * HID_ + n] : 0.f;
    W1s[(g * 32 + l) * 16 + 2 * vv]     = (_Float16)a;
    W1s[(g * 32 + l) * 16 + 2 * vv + 1] = (_Float16)b;
  }

  // W2 -> 16 groups (kc:4 x nt:4), padded 100x64 -> 128x64
  for (int i = tid; i < 16 * 32 * 8; i += 256) {
    const int g = i >> 8, l = (i >> 3) & 31, vv = i & 7;
    const int kc = g >> 2, nt = g & 3;
    const int n  = nt * 16 + (l & 15);
    const int k0 = kc * 32 + ((vv >> 2) << 4) + ((l >> 4) << 3) + 2 * (vv & 3);
    const float a = (k0     < HID_) ? W2[k0 * V_ + n]       : 0.f;
    const float b = (k0 + 1 < HID_) ? W2[(k0 + 1) * V_ + n] : 0.f;
    W2s[(g * 32 + l) * 16 + 2 * vv]     = (_Float16)a;
    W2s[(g * 32 + l) * 16 + 2 * vv + 1] = (_Float16)b;
  }

  for (int i = tid; i < 128; i += 256) b1p[i] = (i < HID_) ? b1[i] : 0.f;
  if (tid < 64) b2p[tid] = b2[tid];
}

// =====================================================================================
// Kernel B: sequential RNN scan. 16 blocks x 4 waves; block owns 16 batch rows,
// wave wv owns N-tile wv of the padded 16x64 h-state.
//   h_t = tanh(Exp[inputs[:,t]] + h_{t-1} @ Wh)    (WMMA C initialized with the gather)
// Two-deep software pipeline:
//   accN   = xp gather for step t   (loads issued at the top of iteration t-1)
//   idxReg = raw token index for t+1 (loaded at iteration t-1)
// At the top of iteration t we immediately issue the t+1 gather and the t+2 index
// load; being real memory ops they cannot sink below the "memory"-clobbered barrier,
// so they overlap WMMA + tanh + the LDS layout round trip.
// =====================================================================================
__global__ __launch_bounds__(128) void rnn_scan(const int*      __restrict__ inputs,
                                                const float*    __restrict__ Exp,
                                                const _Float16* __restrict__ Whs,
                                                _Float16*       __restrict__ hsg)
{
  __shared__ _Float16 hlds[2][16 * 64];

  const int tid  = threadIdx.x;
  const int lane = tid & 31;
  const int wv   = tid >> 5;        // this wave's N-tile (0..3)
  const int hi   = lane >> 4;
  const int n0   = lane & 15;
  const int rb   = blockIdx.x * 16; // batch-row base
  const int* myrow = inputs + (rb + n0) * T_;

  // Wh B-layout tiles for this wave, resident in VGPRs for the whole scan.
  v16h bh0, bh1;
  {
    const _Float16* p0 = Whs + ((0 * 4 + wv) * 32 + lane) * 16;
    const _Float16* p1 = Whs + ((1 * 4 + wv) * 32 + lane) * 16;
    bh0 = concat8(*(const v8h*)p0, *(const v8h*)(p0 + 8));
    bh1 = concat8(*(const v8h*)p1, *(const v8h*)(p1 + 8));
  }

  // h0 = 0 in f16 A-layout
  const v8h z8 = {0, 0, 0, 0, 0, 0, 0, 0};
  v16h a0 = concat8(z8, z8);
  v16h a1 = concat8(z8, z8);

  // prologue: gather xp for t=0, prefetch raw index for t=1
  v8f accN;
  {
    const int idx0 = myrow[0];
#pragma unroll
    for (int g = 0; g < 8; ++g) {
      const int M   = g + 8 * hi;
      const int row = __shfl(idx0, M, 32);            // ds_bpermute broadcast
      accN[g] = Exp[row * 64 + wv * 16 + n0];
    }
  }
  int idxReg = myrow[1];

  for (int t = 0; t < T_; ++t) {
    // (1) issue step-(t+1) gather NOW: idxReg resolved an iteration ago, so the
    //     bpermutes have no load-wait in front and the 8 Exp loads go out first.
    v8f accF;
#pragma unroll
    for (int g = 0; g < 8; ++g) {
      const int M   = g + 8 * hi;
      const int row = __shfl(idxReg, M, 32);
      accF[g] = Exp[row * 64 + wv * 16 + n0];
    }
    // (2) issue raw index load for step t+2 (consumed next iteration)
    {
      const int t3 = (t + 2 < T_) ? t + 2 : T_ - 1;   // uniform, branch-free
      idxReg = myrow[t3];
    }

    // (3) recurrence matmul for step t (C initialized with last iteration's gather)
    v8f acc = accN;
    acc = __builtin_amdgcn_wmma_f32_16x16x32_f16(false, a0, false, bh0,
                                                 (short)0, acc, false, false);
    acc = __builtin_amdgcn_wmma_f32_16x16x32_f16(false, a1, false, bh1,
                                                 (short)0, acc, false, false);

    // (4) tanh (hardware v_tanh_f32), write this wave's 16x16 column into buffer t&1
    _Float16* wbuf = hlds[t & 1];
#pragma unroll
    for (int g = 0; g < 8; ++g) {
      const int M = g + 8 * hi;
      wbuf[M * 64 + wv * 16 + n0] = (_Float16)fast_tanh(acc[g]);
    }
    lds_barrier();   // s_wait_dscnt 0 + s_barrier_signal/-wait -1 (loads keep flying)

    // (5) rebuild f16 A-layout for the next step; ping-pong buffer -> single barrier
    const _Float16* rp = wbuf + n0 * 64 + hi * 8;
    v8h p0 = *(const v8h*)(rp);        // K = hi*8      .. +7
    v8h p1 = *(const v8h*)(rp + 16);   // K = 16 + hi*8 .. +7
    v8h p2 = *(const v8h*)(rp + 32);   // K = 32 + hi*8 .. +7
    v8h p3 = *(const v8h*)(rp + 48);   // K = 48 + hi*8 .. +7
    a0 = concat8(p0, p1);
    a1 = concat8(p2, p3);

    // (6) stream h_t to the global hs buffer (waves 0/1 store one 32-col chunk each)
    _Float16* dst = hsg + ((rb + n0) * T_ + t) * 64 + hi * 8;
    if (wv == 0) {
      *(v8h*)(dst)      = p0;
      *(v8h*)(dst + 16) = p1;
    } else if (wv == 1) {
      *(v8h*)(dst + 32) = p2;
      *(v8h*)(dst + 48) = p3;
    }

    accN = accF;   // rotate the pipeline
  }
}

// =====================================================================================
// Kernel C: parallel MLP head over all B*T rows.
//   hidden = relu(h @ W1 + b1)   [16x128 per tile, f16]
//   logits = hidden @ W2 + b2    [16x64  per tile, f32 out]
// 512 blocks x 8 waves, 8 tiles per wave -> 32768 tiles.
// =====================================================================================
__global__ __launch_bounds__(256) void mlp_head(const _Float16* __restrict__ hsg,
                                                const _Float16* __restrict__ W1s,
                                                const float*    __restrict__ b1p,
                                                const float*    __restrict__ b2p,
                                                const _Float16* __restrict__ W2s,
                                                float*          __restrict__ out)
{
  __shared__ _Float16 w1l[16 * 32 * 16];   // W1 swizzled, 16 KiB
  __shared__ float    bl[192];             // b1 [0..127], b2 [128..191]
  __shared__ _Float16 hid[8 * 16 * 128];   // per-wave hidden tiles, 32 KiB

  const int tid  = threadIdx.x;
  const int lane = tid & 31;
  const int wv   = tid >> 5;
  const int hi   = lane >> 4;
  const int n0   = lane & 15;

  {
    const i4* src = (const i4*)W1s;
    i4* dst = (i4*)w1l;
    for (int i = tid; i < 1024; i += 256) dst[i] = src[i];
    for (int i = tid; i < 192; i += 256) bl[i] = (i < 128) ? b1p[i] : b2p[i - 128];
  }
  __syncthreads();

  _Float16* myhid = hid + wv * (16 * 128);
  const int waveGlobal = blockIdx.x * 8 + wv;

  for (int it = 0; it < 8; ++it) {
    const int tile = waveGlobal * 8 + it;
    const int row0 = tile * 16;

    // prefetch next tile's activations (global_prefetch_b8; speculative, fault-free)
    __builtin_prefetch(hsg + (long)(row0 + 16 + n0) * 64, 0, 1);

    // load h tile (16x64 f16, A-layout)
    const _Float16* hrow = hsg + (row0 + n0) * 64 + hi * 8;
    const v16h ha0 = concat8(*(const v8h*)(hrow),      *(const v8h*)(hrow + 16));
    const v16h ha1 = concat8(*(const v8h*)(hrow + 32), *(const v8h*)(hrow + 48));

    // hidden = relu(h @ W1 + b1)
#pragma unroll
    for (int nt = 0; nt < 8; ++nt) {
      const float bv = bl[nt * 16 + n0];
      v8f acc;
#pragma unroll
      for (int g = 0; g < 8; ++g) acc[g] = bv;
      {
        const _Float16* p0 = w1l + ((0 * 8 + nt) * 32 + lane) * 16;
        const v16h w0 = concat8(*(const v8h*)p0, *(const v8h*)(p0 + 8));
        acc = __builtin_amdgcn_wmma_f32_16x16x32_f16(false, ha0, false, w0,
                                                     (short)0, acc, false, false);
        const _Float16* p1 = w1l + ((1 * 8 + nt) * 32 + lane) * 16;
        const v16h w1 = concat8(*(const v8h*)p1, *(const v8h*)(p1 + 8));
        acc = __builtin_amdgcn_wmma_f32_16x16x32_f16(false, ha1, false, w1,
                                                     (short)0, acc, false, false);
      }
#pragma unroll
      for (int g = 0; g < 8; ++g) {
        const float v = acc[g];
        myhid[(g + 8 * hi) * 128 + nt * 16 + n0] = (_Float16)(v > 0.f ? v : 0.f);
      }
    }

    // rebuild hidden in A-layout (same-wave LDS RAW; compiler inserts s_wait_dscnt)
    const _Float16* rp = myhid + n0 * 128 + hi * 8;
    const v16h ah0 = concat8(*(const v8h*)(rp),      *(const v8h*)(rp + 16));
    const v16h ah1 = concat8(*(const v8h*)(rp + 32), *(const v8h*)(rp + 48));
    const v16h ah2 = concat8(*(const v8h*)(rp + 64), *(const v8h*)(rp + 80));
    const v16h ah3 = concat8(*(const v8h*)(rp + 96), *(const v8h*)(rp + 112));

    // logits = hidden @ W2 + b2  (W2 tiles streamed from L2-resident swizzle buffer)
#pragma unroll
    for (int nt = 0; nt < 4; ++nt) {
      const float bv = bl[128 + nt * 16 + n0];
      v8f acc;
#pragma unroll
      for (int g = 0; g < 8; ++g) acc[g] = bv;
      {
        const _Float16* q0 = W2s + ((0 * 4 + nt) * 32 + lane) * 16;
        const _Float16* q1 = W2s + ((1 * 4 + nt) * 32 + lane) * 16;
        const _Float16* q2 = W2s + ((2 * 4 + nt) * 32 + lane) * 16;
        const _Float16* q3 = W2s + ((3 * 4 + nt) * 32 + lane) * 16;
        const v16h w0 = concat8(*(const v8h*)q0, *(const v8h*)(q0 + 8));
        acc = __builtin_amdgcn_wmma_f32_16x16x32_f16(false, ah0, false, w0,
                                                     (short)0, acc, false, false);
        const v16h w1 = concat8(*(const v8h*)q1, *(const v8h*)(q1 + 8));
        acc = __builtin_amdgcn_wmma_f32_16x16x32_f16(false, ah1, false, w1,
                                                     (short)0, acc, false, false);
        const v16h w2 = concat8(*(const v8h*)q2, *(const v8h*)(q2 + 8));
        acc = __builtin_amdgcn_wmma_f32_16x16x32_f16(false, ah2, false, w2,
                                                     (short)0, acc, false, false);
        const v16h w3 = concat8(*(const v8h*)q3, *(const v8h*)(q3 + 8));
        acc = __builtin_amdgcn_wmma_f32_16x16x32_f16(false, ah3, false, w3,
                                                     (short)0, acc, false, false);
      }
#pragma unroll
      for (int g = 0; g < 8; ++g)
        out[(long)(row0 + g + 8 * hi) * 64 + nt * 16 + n0] = acc[g];
    }
  }
}

// =====================================================================================
extern "C" void kernel_launch(void* const* d_in, const int* in_sizes, int n_in,
                              void* d_out, int out_size, void* d_ws, size_t ws_size,
                              hipStream_t stream)
{
  (void)in_sizes; (void)n_in; (void)out_size; (void)ws_size;
  const int*   inputs = (const int*)  d_in[0];
  const float* embed  = (const float*)d_in[1];
  const float* Wx     = (const float*)d_in[2];
  const float* Wh     = (const float*)d_in[3];
  const float* b_rnn  = (const float*)d_in[4];
  const float* W1     = (const float*)d_in[5];
  const float* b1     = (const float*)d_in[6];
  const float* W2     = (const float*)d_in[7];
  const float* b2     = (const float*)d_in[8];
  unsigned char* ws = (unsigned char*)d_ws;

  prep_weights<<<1, 256, 0, stream>>>(embed, Wx, Wh, b_rnn, W1, b1, W2, b2, ws);
  rnn_scan<<<16, 128, 0, stream>>>(inputs,
                                   (const float*)(ws + WS_EXP),
                                   (const _Float16*)(ws + WS_WHS),
                                   (_Float16*)(ws + WS_HS));
  mlp_head<<<512, 256, 0, stream>>>((const _Float16*)(ws + WS_HS),
                                    (const _Float16*)(ws + WS_W1S),
                                    (const float*)(ws + WS_B1),
                                    (const float*)(ws + WS_B2),
                                    (const _Float16*)(ws + WS_W2S),
                                    (float*)d_out);
}
